// FlashAttention_81939386073748
// MI455X (gfx1250) — compile-verified, matrix operand order checked by reading, not run on hardware
//
#include <hip/hip_runtime.h>
#include <hip/hip_bf16.h>
#include <math.h>

// ---------------------------------------------------------------------------
// Block-causal attention module for MI455X (gfx1250), wave32 + WMMA.
//   qkv  = f16( (x @ Wqkv + bqkv) * colscale )   (WMMA GEMM, f16 output, Q pre-scaled)
//   attn = per-(h,i-block) loop, f16 output      (WMMA scores + WMMA P@V, async K/Q loads)
//   out  = attn @ Wout + bout                    (WMMA GEMM, async f16 A tiles)
// ---------------------------------------------------------------------------

typedef __attribute__((ext_vector_type(16))) _Float16 v16h;
typedef __attribute__((ext_vector_type(8)))  _Float16 v8h;
typedef __attribute__((ext_vector_type(8)))  float    v8f;
typedef int v4i __attribute__((vector_size(16)));   // param type of async builtin

#define S_LEN   2048
#define DIM     2048
#define NHEAD   16
#define HDIM    128
#define BS      128
#define NB      16
#define QKV_LD  6144          // 3*NHEAD*HDIM
#define INV_SQRT_D 0.08838834764831845f   // 1/sqrt(128)

#define TK      64            // GEMM K-chunk depth
#define LDT     72            // padded LDS row stride (halves): 144B, 16B-aligned
#define LDA     136           // attention LDS row stride (halves): 272B, 16B-aligned

// ---------------------------------------------------------------------------
// 16B global -> LDS copy: async DMA path (ASYNCcnt) when available.
// ---------------------------------------------------------------------------
__device__ __forceinline__ void copy16_g2l(const _Float16* g, _Float16* l) {
#if __has_builtin(__builtin_amdgcn_global_load_async_to_lds_b128)
  __builtin_amdgcn_global_load_async_to_lds_b128(
      (v4i*)(void*)g, (v4i*)(void*)l, 0, 0);
#else
  *(v8h*)l = *(const v8h*)g;
#endif
}
__device__ __forceinline__ void wait_async() {
#if __has_builtin(__builtin_amdgcn_global_load_async_to_lds_b128)
#if __has_builtin(__builtin_amdgcn_s_wait_asynccnt)
  __builtin_amdgcn_s_wait_asynccnt(0);
#else
  asm volatile("s_wait_asynccnt 0x0" ::: "memory");
#endif
#endif
}

// ---------------------------------------------------------------------------
// 16x32 f16 fragment (A-layout) from LDS, row-major, stride `ld` halves.
//   lanes 0-15 : row = lane,    K = k0+{0..7, 16..23}
//   lanes 16-31: row = lane-16, K = k0+{8..15, 24..31}
// Two contiguous 8-half groups -> two ds_load_b128 per fragment.
// B fragments use the same mapping on a [n][k]-transposed LDS tile.
// ---------------------------------------------------------------------------
__device__ __forceinline__ v16h load_frag(const _Float16* base, int ld, int k0) {
  const int lane = threadIdx.x & 31;
  const int row  = lane & 15;
  const int kb   = (lane >> 4) << 3;          // 0 or 8
  const _Float16* p = base + row * ld + k0 + kb;
  v8h lo = *(const v8h*)(p);
  v8h hi = *(const v8h*)(p + 16);
  v16h f;
#pragma unroll
  for (int e = 0; e < 8; ++e) { f[e] = lo[e]; f[e + 8] = hi[e]; }
  return f;
}

// ---------------------------------------------------------------------------
// C = cast<TC>( scale(col) * (A[MxK] @ B[KxN] + bias[N]) )
// TA = float  : A staged through registers with f32->f16 conversion.
// TA = f16    : A tile copied with GLOBAL_LOAD_ASYNC_TO_LDS_B128 (no VGPRs).
// 128x128 tile per workgroup, 8 waves x (16 rows x 128 cols), ping-pong LDS.
// cols < qscale_cols get INV_SQRT_D folded into the epilogue (Q pre-scaling).
// ---------------------------------------------------------------------------
template <typename TA, typename TC>
__global__ __launch_bounds__(256) void gemm_wmma(
    const TA* __restrict__ A, const float* __restrict__ B,
    const float* __restrict__ bias, TC* __restrict__ C,
    int M, int N, int K, int qscale_cols)
{
  __shared__ _Float16 As[2][128 * LDT];   // [m][k]
  __shared__ _Float16 Bs[2][128 * LDT];   // [n][k] transposed

  const int tid  = threadIdx.x;
  const int lane = tid & 31;
  const int wave = tid >> 5;
  const int m0   = blockIdx.y * 128;
  const int n0   = blockIdx.x * 128;

  // f32-A loader: row = (tid>>4)+16p, col = (tid&15)*4, 8 passes of float4
  const int arow   = tid >> 4, acol   = (tid & 15) * 4;
  // f16-A async loader: row = (tid>>3)+32p, col = (tid&7)*8, 4 passes of 16B
  const int arow16 = tid >> 3, acol16 = (tid & 7) * 8;
  // B loader: krow = (tid>>5)+8p, col = (tid&31)*4, 8 passes of float4
  const int brow   = tid >> 5, bcol   = (tid & 31) * 4;

  float4 aReg[8], bReg[8];

  auto loadA = [&](int k0, int buf) {
    if constexpr (sizeof(TA) == 2) {        // async DMA straight into LDS
#pragma unroll
      for (int p = 0; p < 4; ++p)
        copy16_g2l((const _Float16*)A + (size_t)(m0 + arow16 + p * 32) * K + k0 + acol16,
                   As[buf] + (arow16 + p * 32) * LDT + acol16);
    } else {                                // stage f32 in registers
      (void)buf;
#pragma unroll
      for (int p = 0; p < 8; ++p)
        aReg[p] = *(const float4*)((const float*)A + (size_t)(m0 + arow + p * 16) * K + k0 + acol);
    }
  };
  auto loadB = [&](int k0) {
#pragma unroll
    for (int p = 0; p < 8; ++p)
      bReg[p] = *(const float4*)(B + (size_t)(k0 + brow + p * 8) * N + n0 + bcol);
  };
  auto storeLDS = [&](int buf) {
    if constexpr (sizeof(TA) == 4) {
#pragma unroll
      for (int p = 0; p < 8; ++p) {
        _Float16* d = As[buf] + (arow + p * 16) * LDT + acol;
        d[0] = (_Float16)aReg[p].x; d[1] = (_Float16)aReg[p].y;
        d[2] = (_Float16)aReg[p].z; d[3] = (_Float16)aReg[p].w;
      }
    }
#pragma unroll
    for (int p = 0; p < 8; ++p) {
      const int kr = brow + p * 8;
      Bs[buf][(bcol + 0) * LDT + kr] = (_Float16)bReg[p].x;
      Bs[buf][(bcol + 1) * LDT + kr] = (_Float16)bReg[p].y;
      Bs[buf][(bcol + 2) * LDT + kr] = (_Float16)bReg[p].z;
      Bs[buf][(bcol + 3) * LDT + kr] = (_Float16)bReg[p].w;
    }
  };

  v8f acc[8] = {};

  loadA(0, 0);
  loadB(0);
  storeLDS(0);
  if constexpr (sizeof(TA) == 2) wait_async();
  __syncthreads();

  const int T = K / TK;
  for (int t = 0; t < T; ++t) {
    const int buf = t & 1;
    if (t + 1 < T) {                       // issue next chunk; overlaps WMMAs
      loadA((t + 1) * TK, buf ^ 1);        // (async lands directly in buf^1)
      loadB((t + 1) * TK);
    }
#pragma unroll
    for (int kc = 0; kc < 2; ++kc) {
      const v16h a = load_frag(As[buf] + wave * 16 * LDT, LDT, kc * 32);
      v16h b[8];
#pragma unroll
      for (int nt = 0; nt < 8; ++nt)
        b[nt] = load_frag(Bs[buf] + nt * 16 * LDT, LDT, kc * 32);
#pragma unroll
      for (int nt = 0; nt < 8; ++nt)
        acc[nt] = __builtin_amdgcn_wmma_f32_16x16x32_f16(
            false, a, false, b[nt], (short)0, acc[nt], false, false);
    }
    if (t + 1 < T) {
      storeLDS(buf ^ 1);                   // other buffer: safe pre-barrier
      if constexpr (sizeof(TA) == 2) wait_async();
    }
    __syncthreads();
  }

  // ---- epilogue: C-layout (VGPR r -> M=r / r+8 per lane half) ----
  const int mloc = (lane >> 4) << 3;
#pragma unroll
  for (int nt = 0; nt < 8; ++nt) {
    const int col = n0 + nt * 16 + (lane & 15);
    const float bv    = bias[col];
    const float scale = (col < qscale_cols) ? INV_SQRT_D : 1.0f;
#pragma unroll
    for (int r = 0; r < 8; ++r) {
      const int row = m0 + wave * 16 + r + mloc;
      C[(size_t)row * N + col] = (TC)((acc[nt][r] + bv) * scale);
    }
  }
}

// ---------------------------------------------------------------------------
// Per-(head, q-block) block attention with per-k-block softmax (faithful to
// the reference: softmax over each 128-wide k block independently, summed
// over j <= i without global renormalization).  qkv is f16, Q pre-scaled.
// ---------------------------------------------------------------------------
__global__ __launch_bounds__(256) void attn_wmma(
    const _Float16* __restrict__ qkv, _Float16* __restrict__ attn_out)
{
  __shared__ _Float16 Qs[128 * LDA];   // [q][d]   (async copy)
  __shared__ _Float16 Ks[128 * LDA];   // [kp][d]  (async copy)
  __shared__ _Float16 Vt[128 * LDA];   // [d][kp]  (manual transpose)
  __shared__ _Float16 Ps[128 * LDA];   // per-wave 16x128 P tiles (A-layout staging)

  const int tid  = threadIdx.x;
  const int lane = tid & 31;
  const int wave = tid >> 5;
  const int iblk = blockIdx.x;         // q block index
  const int h    = blockIdx.y;         // head
  const int qcol = h * HDIM;
  const int kcol = NHEAD * HDIM + h * HDIM;
  const int vcol = 2 * NHEAD * HDIM + h * HDIM;
  const int crow = tid >> 4;           // 0..15  copy row within pass
  const int c8   = (tid & 15) * 8;     // 0..120 copy col (halves, 16B units)

  // ---- Q block: pure 16B async copies (pre-scaled by GEMM1 epilogue) ----
#pragma unroll
  for (int p = 0; p < 8; ++p) {
    const int q = crow + p * 16;
    copy16_g2l(qkv + (size_t)(iblk * BS + q) * QKV_LD + qcol + c8,
               Qs + q * LDA + c8);
  }

  v8f o[8] = {};                       // 16 q-rows x 128 d output per wave
  wait_async();
  __syncthreads();

  for (int j = 0; j <= iblk; ++j) {
    // ---- K_j: async 16B copies; V_j: load + transposed scatter ----
#pragma unroll
    for (int p = 0; p < 8; ++p) {
      const int kp = crow + p * 16;
      const size_t rowoff = (size_t)(j * BS + kp) * QKV_LD;
      copy16_g2l(qkv + rowoff + kcol + c8, Ks + kp * LDA + c8);
      const v8h vv = *(const v8h*)(qkv + rowoff + vcol + c8);
#pragma unroll
      for (int u = 0; u < 8; ++u)
        Vt[(c8 + u) * LDA + kp] = vv[u];
    }
    wait_async();
    __syncthreads();

    if (j + 1 <= iblk) {               // hint next K/V rows (global_prefetch_b8)
      const size_t nxt = (size_t)((j + 1) * BS + crow) * QKV_LD;
      __builtin_prefetch(qkv + nxt + kcol + c8, 0, 1);
      __builtin_prefetch(qkv + nxt + vcol + c8, 0, 1);
    }

    // ---- scores S = Q @ K^T : 16 q-rows x 128 kp per wave ----
    v8f s[8] = {};
#pragma unroll
    for (int kc = 0; kc < 4; ++kc) {
      const v16h a = load_frag(Qs + wave * 16 * LDA, LDA, kc * 32);
      v16h b[8];
#pragma unroll
      for (int nt = 0; nt < 8; ++nt)
        b[nt] = load_frag(Ks + nt * 16 * LDA, LDA, kc * 32);
#pragma unroll
      for (int nt = 0; nt < 8; ++nt)
        s[nt] = __builtin_amdgcn_wmma_f32_16x16x32_f16(
            false, a, false, b[nt], (short)0, s[nt], false, false);
    }

    const int nbase = lane & 15;
    const int mloc  = (lane >> 4) << 3;

    // ---- causal mask inside the diagonal block ----
    if (j == iblk) {
#pragma unroll
      for (int nt = 0; nt < 8; ++nt) {
        const int kpi = nt * 16 + nbase;
#pragma unroll
        for (int r = 0; r < 8; ++r) {
          const int qi = wave * 16 + r + mloc;
          if (kpi > qi) s[nt][r] = -1e10f;
        }
      }
    }

    // ---- per-k-block softmax over 128 kp; rows live on 16-lane halves ----
#pragma unroll
    for (int r = 0; r < 8; ++r) {
      float mx = s[0][r];
#pragma unroll
      for (int nt = 1; nt < 8; ++nt) mx = fmaxf(mx, s[nt][r]);
#pragma unroll
      for (int off = 1; off < 16; off <<= 1) mx = fmaxf(mx, __shfl_xor(mx, off, 32));
      float sum = 0.f;
#pragma unroll
      for (int nt = 0; nt < 8; ++nt) {
        const float e = __expf(s[nt][r] - mx);
        s[nt][r] = e;
        sum += e;
      }
#pragma unroll
      for (int off = 1; off < 16; off <<= 1) sum += __shfl_xor(sum, off, 32);
      const float inv = 1.0f / sum;
#pragma unroll
      for (int nt = 0; nt < 8; ++nt) s[nt][r] *= inv;
    }

    // ---- stage P (C-layout -> A-layout via per-wave LDS region) ----
    {
      _Float16* Pw = Ps + wave * 16 * LDA;
#pragma unroll
      for (int nt = 0; nt < 8; ++nt)
#pragma unroll
        for (int r = 0; r < 8; ++r)
          Pw[(r + mloc) * LDA + nt * 16 + nbase] = (_Float16)s[nt][r];
    }

    // ---- O += P @ V_j  (contraction over kp) ----
#pragma unroll
    for (int kc = 0; kc < 4; ++kc) {
      const v16h a = load_frag(Ps + wave * 16 * LDA, LDA, kc * 32);
      v16h b[8];
#pragma unroll
      for (int nt = 0; nt < 8; ++nt)
        b[nt] = load_frag(Vt + nt * 16 * LDA, LDA, kc * 32);
#pragma unroll
      for (int nt = 0; nt < 8; ++nt)
        o[nt] = __builtin_amdgcn_wmma_f32_16x16x32_f16(
            false, a, false, b[nt], (short)0, o[nt], false, false);
    }
    __syncthreads();   // all waves done with Ks/Vt before next j reload
  }

  // ---- write O (f16) in [s, h*D + d] layout ----
  const int nbase = lane & 15;
  const int mloc  = (lane >> 4) << 3;
#pragma unroll
  for (int nt = 0; nt < 8; ++nt) {
    const int col = h * HDIM + nt * 16 + nbase;
#pragma unroll
    for (int r = 0; r < 8; ++r) {
      const int row = iblk * BS + wave * 16 + r + mloc;
      attn_out[(size_t)row * (NHEAD * HDIM) + col] = (_Float16)o[nt][r];
    }
  }
}

// ---------------------------------------------------------------------------
extern "C" void kernel_launch(void* const* d_in, const int* in_sizes, int n_in,
                              void* d_out, int out_size, void* d_ws, size_t ws_size,
                              hipStream_t stream) {
  const float* x    = (const float*)d_in[0];   // [1,2048,2048]
  const float* Wqkv = (const float*)d_in[1];   // [2048,6144]
  const float* bqkv = (const float*)d_in[2];   // [6144]
  const float* Wout = (const float*)d_in[3];   // [2048,2048]
  const float* bout = (const float*)d_in[4];   // [2048]
  float* out = (float*)d_out;                  // [1,2048,2048]

  // workspace (f16): qkv [S,6144] then attn [S,2048]  (~33.5 MB total)
  _Float16* qkv  = (_Float16*)d_ws;
  _Float16* attn = qkv + (size_t)S_LEN * QKV_LD;

  // 1) qkv = f16( (x @ Wqkv + bqkv) * [q-cols scaled by 1/sqrt(D)] )
  gemm_wmma<float, _Float16><<<dim3(QKV_LD / 128, S_LEN / 128), 256, 0, stream>>>(
      x, Wqkv, bqkv, qkv, S_LEN, QKV_LD, DIM, NHEAD * HDIM);

  // 2) block-causal attention, f16 output in [S, H*D]
  attn_wmma<<<dim3(NB, NHEAD), 256, 0, stream>>>(qkv, attn);

  // 3) out = attn @ Wout + bout   (f16 A tiles via async LDS copies)
  gemm_wmma<_Float16, float><<<dim3(DIM / 128, S_LEN / 128), 256, 0, stream>>>(
      attn, Wout, bout, out, S_LEN, DIM, DIM, 0);
}
